// SEMV8Model_68161130987923
// MI455X (gfx1250) — matrix operand order repaired and checked by reading, compiled, not run on hardware
//
#include <hip/hip_runtime.h>
#include <cstdint>
#include <cstddef>

// ---------------------------------------------------------------------------
// SEMV8 complex-Mamba model for MI455X (gfx1250, wave32, WMMA).
// All large GEMMs -> v_wmma_f32_16x16x32_bf16 with LDS-staged tiles.
//   * A tiles staged via GLOBAL_LOAD_ASYNC_TO_LDS_B128 (ASYNCcnt path)
//   * B tiles staged transposed so fragment reads are ds_load_2addr_b64
// Sequential SSM scan -> one wave per (b,d), lane-parallel over N=64 states.
// Final logits -> fused dual GEMM (real+imag share B tile), writes log|amp|^2.
// ---------------------------------------------------------------------------

#define Vv    32000
#define Dd    1024
#define TWO_D 2048
#define Nn    64
#define Ll    4
#define KCc   4
#define KLl   4
#define Bb    2
#define Ss    1024
#define Mm    (Bb * Ss)          // 2048
#define SCALE_F  0.35355339059327373f   // 1/sqrt(2*L)
#define GAMMA_F  0.01f
#define COND_TH_F 100.0f

typedef __attribute__((ext_vector_type(16))) __bf16 v16bf;
typedef __attribute__((ext_vector_type(8)))  float  v8f;

union FragBF { v16bf v; unsigned u[8]; unsigned short h[16]; };
union FragF  { v8f  v; float f[8]; };

__device__ __forceinline__ unsigned short f2bf(float f) {
  union { float f; unsigned u; } v; v.f = f;
  unsigned u = v.u;
  u += 0x7FFFu + ((u >> 16) & 1u);         // round-to-nearest-even
  return (unsigned short)(u >> 16);
}

// gfx1250 async data mover: move 16B/lane global -> LDS without VGPR bounce.
// LDS destination = low 32 bits of the generic pointer (flat aperture -> LDS).
__device__ __forceinline__ void async_load_b128(const void* gptr, void* lptr) {
  unsigned lds = (unsigned)(size_t)lptr;
  unsigned long long ga = (unsigned long long)(size_t)gptr;
  asm volatile("global_load_async_to_lds_b128 %0, %1, off"
               :: "v"(lds), "v"(ga) : "memory");
}
__device__ __forceinline__ void wait_async() {
  asm volatile("s_wait_asynccnt 0x0" ::: "memory");
}

// ------------------------------ weight prep --------------------------------
__global__ void k_cvt_bf16(const float* __restrict__ src,
                           unsigned short* __restrict__ dst, int n) {
  int i = blockIdx.x * blockDim.x + threadIdx.x;
  if (i < n) dst[i] = f2bf(src[i]);
}

// dst[c*R + r] = bf16(src[r*C + c])   (R x C) -> (C x R)
__global__ void k_transpose_bf16(const float* __restrict__ src,
                                 unsigned short* __restrict__ dst, int R, int C) {
  int i = blockIdx.x * blockDim.x + threadIdx.x;
  if (i >= R * C) return;
  int r = i / C, c = i % C;
  dst[(size_t)c * R + r] = f2bf(src[(size_t)r * C + c]);
}

// ------------------------------- embedding ---------------------------------
__global__ void k_embed(const int* __restrict__ tok,
                        const float* __restrict__ er, const float* __restrict__ ei,
                        float* __restrict__ pr, float* __restrict__ pi,
                        unsigned short* __restrict__ abr, unsigned short* __restrict__ abi) {
  int i = blockIdx.x * blockDim.x + threadIdx.x;
  if (i >= Mm * Dd) return;
  int m = i / Dd, d = i % Dd;
  int t = tok[m];
  float r = er[(size_t)t * Dd + d];
  float im = ei[(size_t)t * Dd + d];
  pr[i] = r; pi[i] = im;
  abr[i] = f2bf(r); abi[i] = f2bf(im);
}

// -------------------- bf16 WMMA GEMM: C(MxN)=A(MxK)B(KxN) ------------------
#define BM  64
#define BN  128
#define BK  32
#define AST 40    // A LDS row stride (halfwords): 80B rows keep 16B alignment
#define BST 36

__global__ __launch_bounds__(256)
void k_gemm(const unsigned short* __restrict__ A,
            const unsigned short* __restrict__ Bw,
            float* __restrict__ C, int Mr, int Nr, int Kr) {
  __shared__ __align__(16) unsigned short As[BM * AST];
  __shared__ __align__(16) unsigned short Bs[BN * BST];
  const int tid  = threadIdx.x;
  const int wave = tid >> 5, lane = tid & 31;
  const int wm = wave >> 2, wn = wave & 3;      // 2 x 4 wave grid
  const int mblk = blockIdx.y * BM;
  const int nblk = blockIdx.x * BN;
  const int lm = lane & 15, kh = lane >> 4;

  v8f zero = {};
  v8f acc[2][2]; acc[0][0] = zero; acc[0][1] = zero; acc[1][0] = zero; acc[1][1] = zero;

  const int am = tid >> 2;                      // 0..63 A row
  const int ak = (tid & 3) * 8;                 // k sub-offset
  const int bk = tid >> 3;                      // 0..31 B k-row
  const int bn = (tid & 7) * 16;                // B n-offset
  const bool nfull = (nblk + BN) <= Nr;

  for (int k0 = 0; k0 < Kr; k0 += BK) {
    // A tile (64x32): async DMA straight into LDS, 16B per lane
    async_load_b128(A + (size_t)(mblk + am) * Kr + (k0 + ak), &As[am * AST + ak]);

    // B tile (32x128) transposed -> Bs[n][k] so frag reads are k-contiguous
    if (nfull) {
      const unsigned short* bp = Bw + (size_t)(k0 + bk) * Nr + (nblk + bn);
      uint4 b0 = *(const uint4*)bp;
      uint4 b1 = *(const uint4*)(bp + 8);
      unsigned short tmp[16];
      *(uint4*)&tmp[0] = b0; *(uint4*)&tmp[8] = b1;
#pragma unroll
      for (int e = 0; e < 16; ++e) Bs[(bn + e) * BST + bk] = tmp[e];
    } else {
#pragma unroll
      for (int e = 0; e < 16; ++e) {
        int gn = nblk + bn + e;
        Bs[(bn + e) * BST + bk] =
            (gn < Nr) ? Bw[(size_t)(k0 + bk) * Nr + gn] : (unsigned short)0;
      }
    }
    wait_async();
    __syncthreads();

    FragBF a[2], b[2];
#pragma unroll
    for (int mi = 0; mi < 2; ++mi) {
      int row = wm * 32 + mi * 16 + lm;
#pragma unroll
      for (int v = 0; v < 8; ++v) {            // ISA A layout: K pairs split by lane half
        int koff = ((v < 4) ? 2 * v : 8 + 2 * v) + 8 * kh;
        a[mi].u[v] = *(const unsigned*)&As[row * AST + koff];
      }
    }
#pragma unroll
    for (int ni = 0; ni < 2; ++ni) {
      int col = wn * 32 + ni * 16 + lm;
#pragma unroll
      for (int v = 0; v < 8; ++v)
        b[ni].u[v] = *(const unsigned*)&Bs[col * BST + ((v < 4) ? 2 * v : 8 + 2 * v) + 8 * kh];
    }
#pragma unroll
    for (int mi = 0; mi < 2; ++mi)
#pragma unroll
      for (int ni = 0; ni < 2; ++ni)
        acc[mi][ni] = __builtin_amdgcn_wmma_f32_16x16x32_bf16(
            false, a[mi].v, false, b[ni].v, (short)0, acc[mi][ni], false, false);
    __syncthreads();
  }

#pragma unroll
  for (int mi = 0; mi < 2; ++mi)
#pragma unroll
    for (int ni = 0; ni < 2; ++ni) {
      FragF cf; cf.v = acc[mi][ni];
      int col = nblk + wn * 32 + ni * 16 + lm;
      if (col < Nr) {
#pragma unroll
        for (int j = 0; j < 8; ++j) {
          int row = mblk + wm * 32 + mi * 16 + j + 8 * kh;   // ISA C/D layout
          C[(size_t)row * Nr + col] = cf.f[j];
        }
      }
    }
}

// --------------- causal depthwise conv + complex SiLU + dt -----------------
__global__ void k_conv(const float* __restrict__ xzr, const float* __restrict__ xzi,
                       const float* __restrict__ ck,
                       const float* __restrict__ dts, const float* __restrict__ dtb,
                       float* __restrict__ x1r, float* __restrict__ x1i,
                       unsigned short* __restrict__ x1rb, unsigned short* __restrict__ x1ib,
                       float* __restrict__ dt) {
  int i = blockIdx.x * blockDim.x + threadIdx.x;
  if (i >= Mm * Dd) return;
  int m = i / Dd, d = i % Dd;
  int b = m / Ss, s = m % Ss;
  float ar = 0.f, ai = 0.f;
#pragma unroll
  for (int j = 0; j < KCc; ++j) {
    int ss = s - (KCc - 1) + j;
    if (ss >= 0) {
      size_t q = (size_t)(b * Ss + ss) * TWO_D + d;
      float kw = ck[d * KCc + j];
      ar += kw * xzr[q];
      ai += kw * xzi[q];
    }
  }
  float mag = sqrtf(ar * ar + ai * ai);
  float sg  = 1.f / (1.f + expf(-mag));
  ar *= sg; ai *= sg;
  float p  = ar * dts[d] + dtb[d];
  float sp = (p > 20.f) ? p : log1pf(expf(p));
  x1r[i] = ar; x1i[i] = ai;
  x1rb[i] = f2bf(ar); x1ib[i] = f2bf(ai);
  dt[i] = sp;
}

// ---------------------- selective scan: wave per (b,d) ---------------------
__global__ __launch_bounds__(256)
void k_scan(const float* __restrict__ dt,
            const float* __restrict__ x1r, const float* __restrict__ x1i,
            const float* __restrict__ Bcr, const float* __restrict__ Bci,
            const float* __restrict__ Ccr, const float* __restrict__ Cci,
            const float* __restrict__ Alog,
            float* __restrict__ yr, float* __restrict__ yi) {
  int wave = threadIdx.x >> 5, lane = threadIdx.x & 31;
  int idx = blockIdx.x * 8 + wave;              // (b,d)
  int b = idx >> 10, d = idx & (Dd - 1);
  float A0 = -expf(Alog[d * Nn + lane]);
  float A1 = -expf(Alog[d * Nn + lane + 32]);
  float h0r = 0.f, h0i = 0.f, h1r = 0.f, h1i = 0.f;
  for (int t = 0; t < Ss; ++t) {
    int m = b * Ss + t;
    float dtv = dt[(size_t)m * Dd + d];
    float xr = x1r[(size_t)m * Dd + d];
    float xi = x1i[(size_t)m * Dd + d];
    float br0 = Bcr[m * Nn + lane],      bi0 = Bci[m * Nn + lane];
    float br1 = Bcr[m * Nn + lane + 32], bi1 = Bci[m * Nn + lane + 32];
    float cr0 = Ccr[m * Nn + lane],      ci0 = Cci[m * Nn + lane];
    float cr1 = Ccr[m * Nn + lane + 32], ci1 = Cci[m * Nn + lane + 32];
    float d0 = expf(dtv * A0), d1 = expf(dtv * A1);
    float ur = dtv * xr, ui = dtv * xi;
    h0r = d0 * h0r + ur * br0 - ui * bi0;
    h0i = d0 * h0i + ur * bi0 + ui * br0;
    h1r = d1 * h1r + ur * br1 - ui * bi1;
    h1i = d1 * h1i + ur * bi1 + ui * br1;
    float sr = h0r * cr0 - h0i * ci0 + h1r * cr1 - h1i * ci1;
    float si = h0r * ci0 + h0i * cr0 + h1r * ci1 + h1i * cr1;
#pragma unroll
    for (int o = 16; o; o >>= 1) {
      sr += __shfl_xor(sr, o, 32);
      si += __shfl_xor(si, o, 32);
    }
    if (lane == 0) {
      yr[(size_t)m * Dd + d] = sr;
      yi[(size_t)m * Dd + d] = si;
    }
  }
}

// -------------------- skip + gate y = (ys + Dk*x1) * csilu(z) --------------
__global__ void k_gate(const float* __restrict__ ysr, const float* __restrict__ ysi,
                       const float* __restrict__ x1r, const float* __restrict__ x1i,
                       const float* __restrict__ xzr, const float* __restrict__ xzi,
                       const float* __restrict__ Dk,
                       unsigned short* __restrict__ ybr, unsigned short* __restrict__ ybi) {
  int i = blockIdx.x * blockDim.x + threadIdx.x;
  if (i >= Mm * Dd) return;
  int m = i / Dd, d = i % Dd;
  float yr = ysr[i] + Dk[d] * x1r[i];
  float yi = ysi[i] + Dk[d] * x1i[i];
  size_t zq = (size_t)m * TWO_D + Dd + d;
  float zr = xzr[zq], zi = xzi[zq];
  float mag = sqrtf(zr * zr + zi * zi);
  float sg  = 1.f / (1.f + expf(-mag));
  float wr = zr * sg, wi = zi * sg;
  float orr = yr * wr - yi * wi;
  float oii = yr * wi + yi * wr;
  ybr[i] = f2bf(orr); ybi[i] = f2bf(oii);
}

// --------------------- Lipschitz projection (K=D dot, KL=4) ----------------
__global__ __launch_bounds__(256)
void k_proj(const float* __restrict__ br, const float* __restrict__ bi,
            const float* __restrict__ Lop,
            float* __restrict__ pr, float* __restrict__ pi) {
  __shared__ float red[256];
  int m = blockIdx.x, tid = threadIdx.x;
  for (int k = 0; k < KLl; ++k) {
    float sr = 0.f, si = 0.f;
    for (int d = tid; d < Dd; d += 256) {
      float lw = Lop[k * Dd + d];
      sr += br[(size_t)m * Dd + d] * lw;
      si += bi[(size_t)m * Dd + d] * lw;
    }
    red[tid] = sr; __syncthreads();
    for (int o = 128; o; o >>= 1) { if (tid < o) red[tid] += red[tid + o]; __syncthreads(); }
    if (tid == 0) pr[m * KLl + k] = red[0];
    __syncthreads();
    red[tid] = si; __syncthreads();
    for (int o = 128; o; o >>= 1) { if (tid < o) red[tid] += red[tid + o]; __syncthreads(); }
    if (tid == 0) pi[m * KLl + k] = red[0];
    __syncthreads();
  }
}

__global__ void k_lupdate(float* __restrict__ br, float* __restrict__ bi,
                          const float* __restrict__ pr, const float* __restrict__ pi,
                          const float* __restrict__ Lop) {
  int i = blockIdx.x * blockDim.x + threadIdx.x;
  if (i >= Mm * Dd) return;
  int m = i / Dd, d = i % Dd;
  float ar = 0.f, ai = 0.f;
#pragma unroll
  for (int k = 0; k < KLl; ++k) {
    float lw = Lop[k * Dd + d];
    ar += pr[m * KLl + k] * lw;
    ai += pi[m * KLl + k] * lw;
  }
  br[i] -= GAMMA_F * ar;
  bi[i] -= GAMMA_F * ai;
}

// ------------------ conditioning diag / phase / residual -------------------
__global__ __launch_bounds__(256)
void k_diag(const float* __restrict__ br, const float* __restrict__ bi,
            float* __restrict__ diag) {
  __shared__ float red[256];
  int blk = blockIdx.x;                   // b*D + d
  int b = blk / Dd, d = blk % Dd;
  int tid = threadIdx.x;
  float s = 0.f;
  for (int t = tid; t < Ss; t += 256) {
    size_t q = (size_t)(b * Ss + t) * Dd + d;
    float r = br[q], im = bi[q];
    s += r * r + im * im;
  }
  red[tid] = s; __syncthreads();
  for (int o = 128; o; o >>= 1) { if (tid < o) red[tid] += red[tid + o]; __syncthreads(); }
  if (tid == 0) diag[blk] = red[0] / (float)Ss;
}

__global__ __launch_bounds__(256)
void k_cond(const float* __restrict__ diag, float* __restrict__ phase) {
  __shared__ float rmax[256], rmin[256];
  int b = blockIdx.x, tid = threadIdx.x;
  float mx = -1e30f, mn = 1e30f;
  for (int d = tid; d < Dd; d += 256) {
    float v = diag[b * Dd + d];
    mx = fmaxf(mx, v); mn = fminf(mn, v);
  }
  rmax[tid] = mx; rmin[tid] = mn; __syncthreads();
  for (int o = 128; o; o >>= 1) {
    if (tid < o) {
      rmax[tid] = fmaxf(rmax[tid], rmax[tid + o]);
      rmin[tid] = fminf(rmin[tid], rmin[tid + o]);
    }
    __syncthreads();
  }
  if (tid == 0) {
    float cond = rmax[0] / (rmin[0] + 1e-8f);
    if (cond > COND_TH_F) { phase[2 * b] = cosf(0.1f); phase[2 * b + 1] = sinf(0.1f); }
    else                  { phase[2 * b] = 1.f;        phase[2 * b + 1] = 0.f; }
  }
}

__global__ void k_resid(float* __restrict__ psir, float* __restrict__ psii,
                        const float* __restrict__ br, const float* __restrict__ bi,
                        const float* __restrict__ phase,
                        unsigned short* __restrict__ abr, unsigned short* __restrict__ abi) {
  int i = blockIdx.x * blockDim.x + threadIdx.x;
  if (i >= Mm * Dd) return;
  int m = i / Dd;
  int b = m / Ss;
  float pr = phase[2 * b], pi = phase[2 * b + 1];
  float nr = br[i] * pr - bi[i] * pi;
  float ni = br[i] * pi + bi[i] * pr;
  float r  = psir[i] + SCALE_F * nr;
  float im = psii[i] + SCALE_F * ni;
  psir[i] = r; psii[i] = im;
  abr[i] = f2bf(r); abi[i] = f2bf(im);
}

// ------------------------------ RMS norm -----------------------------------
__global__ __launch_bounds__(256)
void k_norm(const float* __restrict__ psir, const float* __restrict__ psii,
            const float* __restrict__ nw,
            unsigned short* __restrict__ abr, unsigned short* __restrict__ abi) {
  __shared__ float red[256];
  int m = blockIdx.x, tid = threadIdx.x;
  float s = 0.f;
  for (int d = tid; d < Dd; d += 256) {
    float r = psir[(size_t)m * Dd + d], im = psii[(size_t)m * Dd + d];
    s += r * r + im * im;
  }
  red[tid] = s; __syncthreads();
  for (int o = 128; o; o >>= 1) { if (tid < o) red[tid] += red[tid + o]; __syncthreads(); }
  float inv = 1.f / sqrtf(red[0] / (float)Dd + 1e-8f);
  __syncthreads();
  for (int d = tid; d < Dd; d += 256) {
    float sc = nw[d] * inv;
    abr[(size_t)m * Dd + d] = f2bf(psir[(size_t)m * Dd + d] * sc);
    abi[(size_t)m * Dd + d] = f2bf(psii[(size_t)m * Dd + d] * sc);
  }
}

// -------- fused dual logits GEMM: shares embT tile between real & imag -----
__global__ __launch_bounds__(256)
void k_logits(const unsigned short* __restrict__ Ar, const unsigned short* __restrict__ Ai,
              const unsigned short* __restrict__ Bw, float* __restrict__ out) {
  __shared__ __align__(16) unsigned short Asr[BM * AST];
  __shared__ __align__(16) unsigned short Asi[BM * AST];
  __shared__ __align__(16) unsigned short Bs[BN * BST];
  const int tid  = threadIdx.x;
  const int wave = tid >> 5, lane = tid & 31;
  const int wm = wave >> 2, wn = wave & 3;
  const int mblk = blockIdx.y * BM;
  const int nblk = blockIdx.x * BN;          // V=32000 = 250*128, always full
  const int lm = lane & 15, kh = lane >> 4;

  v8f zero = {};
  v8f accr[2][2], acci[2][2];
  accr[0][0] = zero; accr[0][1] = zero; accr[1][0] = zero; accr[1][1] = zero;
  acci[0][0] = zero; acci[0][1] = zero; acci[1][0] = zero; acci[1][1] = zero;

  const int am = tid >> 2;
  const int ak = (tid & 3) * 8;
  const int bk = tid >> 3;
  const int bn = (tid & 7) * 16;

  for (int k0 = 0; k0 < Dd; k0 += BK) {
    // real + imag A tiles: async DMA into LDS (no transposition needed)
    async_load_b128(Ar + (size_t)(mblk + am) * Dd + (k0 + ak), &Asr[am * AST + ak]);
    async_load_b128(Ai + (size_t)(mblk + am) * Dd + (k0 + ak), &Asi[am * AST + ak]);
    {
      const unsigned short* bp = Bw + (size_t)(k0 + bk) * Vv + (nblk + bn);
      if (k0 + BK < Dd) __builtin_prefetch(bp + (size_t)BK * Vv, 0, 1);
      uint4 b0 = *(const uint4*)bp;
      uint4 b1 = *(const uint4*)(bp + 8);
      unsigned short tmp[16];
      *(uint4*)&tmp[0] = b0; *(uint4*)&tmp[8] = b1;
#pragma unroll
      for (int e = 0; e < 16; ++e) Bs[(bn + e) * BST + bk] = tmp[e];
    }
    wait_async();
    __syncthreads();

    FragBF ar[2], ai2[2], b[2];
#pragma unroll
    for (int mi = 0; mi < 2; ++mi) {
      int row = wm * 32 + mi * 16 + lm;
#pragma unroll
      for (int v = 0; v < 8; ++v) {
        int koff = ((v < 4) ? 2 * v : 8 + 2 * v) + 8 * kh;
        ar[mi].u[v]  = *(const unsigned*)&Asr[row * AST + koff];
        ai2[mi].u[v] = *(const unsigned*)&Asi[row * AST + koff];
      }
    }
#pragma unroll
    for (int ni = 0; ni < 2; ++ni) {
      int col = wn * 32 + ni * 16 + lm;
#pragma unroll
      for (int v = 0; v < 8; ++v)
        b[ni].u[v] = *(const unsigned*)&Bs[col * BST + ((v < 4) ? 2 * v : 8 + 2 * v) + 8 * kh];
    }
#pragma unroll
    for (int mi = 0; mi < 2; ++mi)
#pragma unroll
      for (int ni = 0; ni < 2; ++ni) {
        accr[mi][ni] = __builtin_amdgcn_wmma_f32_16x16x32_bf16(
            false, ar[mi].v, false, b[ni].v, (short)0, accr[mi][ni], false, false);
        acci[mi][ni] = __builtin_amdgcn_wmma_f32_16x16x32_bf16(
            false, ai2[mi].v, false, b[ni].v, (short)0, acci[mi][ni], false, false);
      }
    __syncthreads();
  }

#pragma unroll
  for (int mi = 0; mi < 2; ++mi)
#pragma unroll
    for (int ni = 0; ni < 2; ++ni) {
      FragF cr, ci; cr.v = accr[mi][ni]; ci.v = acci[mi][ni];
      int col = nblk + wn * 32 + ni * 16 + lm;
#pragma unroll
      for (int j = 0; j < 8; ++j) {
        int row = mblk + wm * 32 + mi * 16 + j + 8 * kh;
        float r = cr.f[j], im = ci.f[j];
        out[(size_t)row * Vv + col] = logf(r * r + im * im + 1e-8f);
      }
    }
}

// ===========================================================================
extern "C" void kernel_launch(void* const* d_in, const int* in_sizes, int n_in,
                              void* d_out, int out_size, void* d_ws, size_t ws_size,
                              hipStream_t stream) {
  const int*   tok     = (const int*)  d_in[0];
  const float* emb_r   = (const float*)d_in[1];
  const float* emb_i   = (const float*)d_in[2];
  const float* W_in    = (const float*)d_in[3];
  const float* conv_k  = (const float*)d_in[4];
  const float* A_log   = (const float*)d_in[5];
  const float* dt_sc   = (const float*)d_in[6];
  const float* dt_bi   = (const float*)d_in[7];
  const float* W_B     = (const float*)d_in[8];
  const float* W_C     = (const float*)d_in[9];
  const float* D_skip  = (const float*)d_in[10];
  const float* W_out   = (const float*)d_in[11];
  const float* L_ops   = (const float*)d_in[12];
  const float* norm_w  = (const float*)d_in[13];
  float* out = (float*)d_out;

  // ---- workspace layout (~228 MB) ----
  size_t off = 0;
  auto alloc = [&](size_t bytes) -> char* {
    char* p = (char*)d_ws + off;
    off += (bytes + 255) & ~(size_t)255;
    return p;
  };
  float* psir = (float*)alloc((size_t)Mm * Dd * 4);
  float* psii = (float*)alloc((size_t)Mm * Dd * 4);
  unsigned short* abr = (unsigned short*)alloc((size_t)Mm * Dd * 2);
  unsigned short* abi = (unsigned short*)alloc((size_t)Mm * Dd * 2);
  float* xzr = (float*)alloc((size_t)Mm * TWO_D * 4);
  float* xzi = (float*)alloc((size_t)Mm * TWO_D * 4);
  float* x1r = (float*)alloc((size_t)Mm * Dd * 4);
  float* x1i = (float*)alloc((size_t)Mm * Dd * 4);
  unsigned short* x1rb = (unsigned short*)alloc((size_t)Mm * Dd * 2);
  unsigned short* x1ib = (unsigned short*)alloc((size_t)Mm * Dd * 2);
  float* dta = (float*)alloc((size_t)Mm * Dd * 4);
  float* Bcr = (float*)alloc((size_t)Mm * Nn * 4);
  float* Bci = (float*)alloc((size_t)Mm * Nn * 4);
  float* Ccr = (float*)alloc((size_t)Mm * Nn * 4);
  float* Cci = (float*)alloc((size_t)Mm * Nn * 4);
  float* ysr = (float*)alloc((size_t)Mm * Dd * 4);
  float* ysi = (float*)alloc((size_t)Mm * Dd * 4);
  unsigned short* ybr = (unsigned short*)alloc((size_t)Mm * Dd * 2);
  unsigned short* ybi = (unsigned short*)alloc((size_t)Mm * Dd * 2);
  float* brr = (float*)alloc((size_t)Mm * Dd * 4);
  float* bri = (float*)alloc((size_t)Mm * Dd * 4);
  float* prj = (float*)alloc((size_t)Mm * KLl * 4);
  float* pij = (float*)alloc((size_t)Mm * KLl * 4);
  float* diag = (float*)alloc((size_t)Bb * Dd * 4);
  float* phase = (float*)alloc(2 * Bb * 4);
  unsigned short* Win_bf  = (unsigned short*)alloc((size_t)Ll * Dd * TWO_D * 2);
  unsigned short* Wout_bf = (unsigned short*)alloc((size_t)Ll * Dd * Dd * 2);
  unsigned short* WBt_bf  = (unsigned short*)alloc((size_t)Ll * Dd * Nn * 2);
  unsigned short* WCt_bf  = (unsigned short*)alloc((size_t)Ll * Dd * Nn * 2);
  unsigned short* embT_bf = (unsigned short*)alloc((size_t)Dd * Vv * 2);

  const int TB = 256;
  const int eltMD = Mm * Dd;
  dim3 gMD((eltMD + TB - 1) / TB);

  // ---- weight prep (bf16) ----
  { int n = Ll * Dd * TWO_D; k_cvt_bf16<<<(n + TB - 1) / TB, TB, 0, stream>>>(W_in, Win_bf, n); }
  { int n = Ll * Dd * Dd;    k_cvt_bf16<<<(n + TB - 1) / TB, TB, 0, stream>>>(W_out, Wout_bf, n); }
  for (int l = 0; l < Ll; ++l) {
    int n = Nn * Dd;
    k_transpose_bf16<<<(n + TB - 1) / TB, TB, 0, stream>>>(W_B + (size_t)l * Nn * Dd,
                                                           WBt_bf + (size_t)l * Dd * Nn, Nn, Dd);
    k_transpose_bf16<<<(n + TB - 1) / TB, TB, 0, stream>>>(W_C + (size_t)l * Nn * Dd,
                                                           WCt_bf + (size_t)l * Dd * Nn, Nn, Dd);
  }
  { int n = Vv * Dd; k_transpose_bf16<<<(n + TB - 1) / TB, TB, 0, stream>>>(emb_r, embT_bf, Vv, Dd); }

  // ---- embedding ----
  k_embed<<<gMD, TB, 0, stream>>>(tok, emb_r, emb_i, psir, psii, abr, abi);

  // ---- layers ----
  for (int l = 0; l < Ll; ++l) {
    const unsigned short* Wi = Win_bf + (size_t)l * Dd * TWO_D;
    const unsigned short* Wo = Wout_bf + (size_t)l * Dd * Dd;
    const unsigned short* WBt = WBt_bf + (size_t)l * Dd * Nn;
    const unsigned short* WCt = WCt_bf + (size_t)l * Dd * Nn;
    const float* ckL  = conv_k + (size_t)l * Dd * KCc;
    const float* dtsL = dt_sc + (size_t)l * Dd;
    const float* dtbL = dt_bi + (size_t)l * Dd;
    const float* AL   = A_log + (size_t)l * Dd * Nn;
    const float* DkL  = D_skip + (size_t)l * Dd;
    const float* LopL = L_ops + (size_t)l * KLl * Dd;

    dim3 gIn((TWO_D + BN - 1) / BN, Mm / BM);
    k_gemm<<<gIn, TB, 0, stream>>>(abr, Wi, xzr, Mm, TWO_D, Dd);
    k_gemm<<<gIn, TB, 0, stream>>>(abi, Wi, xzi, Mm, TWO_D, Dd);

    k_conv<<<gMD, TB, 0, stream>>>(xzr, xzi, ckL, dtsL, dtbL, x1r, x1i, x1rb, x1ib, dta);

    dim3 gBC((Nn + BN - 1) / BN, Mm / BM);
    k_gemm<<<gBC, TB, 0, stream>>>(x1rb, WBt, Bcr, Mm, Nn, Dd);
    k_gemm<<<gBC, TB, 0, stream>>>(x1ib, WBt, Bci, Mm, Nn, Dd);
    k_gemm<<<gBC, TB, 0, stream>>>(x1rb, WCt, Ccr, Mm, Nn, Dd);
    k_gemm<<<gBC, TB, 0, stream>>>(x1ib, WCt, Cci, Mm, Nn, Dd);

    k_scan<<<(Bb * Dd) / 8, TB, 0, stream>>>(dta, x1r, x1i, Bcr, Bci, Ccr, Cci, AL, ysr, ysi);

    k_gate<<<gMD, TB, 0, stream>>>(ysr, ysi, x1r, x1i, xzr, xzi, DkL, ybr, ybi);

    dim3 gOut((Dd + BN - 1) / BN, Mm / BM);
    k_gemm<<<gOut, TB, 0, stream>>>(ybr, Wo, brr, Mm, Dd, Dd);
    k_gemm<<<gOut, TB, 0, stream>>>(ybi, Wo, bri, Mm, Dd, Dd);

    k_proj<<<Mm, TB, 0, stream>>>(brr, bri, LopL, prj, pij);
    k_lupdate<<<gMD, TB, 0, stream>>>(brr, bri, prj, pij, LopL);

    k_diag<<<Bb * Dd, TB, 0, stream>>>(brr, bri, diag);
    k_cond<<<Bb, TB, 0, stream>>>(diag, phase);
    k_resid<<<gMD, TB, 0, stream>>>(psir, psii, brr, bri, phase, abr, abi);
  }

  // ---- final norm + fused dual logits GEMM ----
  k_norm<<<Mm, TB, 0, stream>>>(psir, psii, norm_w, abr, abi);
  dim3 gLg(Vv / BN, Mm / BM);   // 250 x 32
  k_logits<<<gLg, TB, 0, stream>>>(abr, abi, embT_bf, out);

  (void)in_sizes; (void)n_in; (void)out_size; (void)ws_size;
}
